// AttentionModule_29094108463808
// MI455X (gfx1250) — matrix operand order repaired and checked by reading, not compiled, not run on hardware
//
#include <hip/hip_runtime.h>
#include <hip/hip_bf16.h>

typedef __attribute__((ext_vector_type(16))) __bf16 v16bf;
typedef __attribute__((ext_vector_type(8)))  float  v8f;
typedef __attribute__((ext_vector_type(4)))  unsigned v4u;
typedef __attribute__((ext_vector_type(8)))  unsigned v8u;

#define B_  4
#define N_  4096
#define M_  4096
#define D_  1024
#define BM  16      // m rows per workgroup
#define BN  64      // n tile
#define NWAVES 8
#define DSLICE 128  // d slice per wave (8 * 128 = 1024)
#define TILE_BYTES (BN * D_ * 2ull)   // global advance per n-tile (128 KB)

__device__ __forceinline__ unsigned f2bf(float f) {
  unsigned u = __float_as_uint(f);
  return (u + 0x7FFFu + ((u >> 16) & 1u)) >> 16;  // round-to-nearest-even
}

__device__ __forceinline__ v8f v8f_zero() {
  v8f z;
#pragma unroll
  for (int i = 0; i < 8; ++i) z[i] = 0.0f;
  return z;
}

union BFrag {
  v16bf v;
  uint4 u4[2];
  unsigned short h[16];
};

// ---------------- phase 0: fp32 -> bf16 for q ----------------
__global__ void cvt_f32_bf16(const float* __restrict__ in,
                             unsigned short* __restrict__ out, int n8) {
  int i = blockIdx.x * blockDim.x + threadIdx.x;
  if (i >= n8) return;
  const float4* p = (const float4*)in + (size_t)i * 2;
  float4 x = p[0], y = p[1];
  uint4 o;
  o.x = f2bf(x.x) | (f2bf(x.y) << 16);
  o.y = f2bf(x.z) | (f2bf(x.w) << 16);
  o.z = f2bf(y.x) | (f2bf(y.y) << 16);
  o.w = f2bf(y.z) | (f2bf(y.w) << 16);
  ((uint4*)out)[i] = o;
}

// Tensor Data Mover: DMA one 64-row x 128-col bf16 tile (16 KB) from global
// (row stride 1024 elements) into LDS, packed row-major (256 B rows).
// 2-D descriptor -> groups 0/1 only (VADDR2/3 = NULL).  Tracked by TENSORcnt.
__device__ __forceinline__ void tdm_load_tile(unsigned lds_addr,
                                              unsigned long long gaddr) {
  v4u g0;
  g0[0] = 1u;                          // count=1, user descriptor
  g0[1] = lds_addr;                    // LDS byte address
  g0[2] = (unsigned)gaddr;             // global_addr[31:0]
  g0[3] = ((unsigned)(gaddr >> 32) & 0x01FFFFFFu) | 0x80000000u; // [56:32]|type=2
  v8u g1;
  g1[0] = 0x00010000u;                 // wg_mask=0, data_size=2B, no flags
  g1[1] = 128u << 16;                  // tensor_dim0 = 128 (lo16)
  g1[2] = 64u  << 16;                  // tensor_dim0 hi | tensor_dim1 = 64 (lo16)
  g1[3] = 128u << 16;                  // tensor_dim1 hi | tile_dim0 = 128
  g1[4] = 64u;                         // tile_dim1 = 64, tile_dim2 = 0
  g1[5] = 1024u;                       // tensor_dim0_stride = 1024 (lo32)
  g1[6] = 0u;                          // stride hi | tensor_dim1_stride lo
  g1[7] = 0u;
  asm volatile("tensor_load_to_lds %0, %1" :: "s"(g0), "s"(g1) : "memory");
}

// ---------------- phase 1: flash attention over the N axis ----------------
// h[b,m,d] = sum_n softmax_n(a[b,m].q[b,n]) * q[b,n,d]
__global__ __launch_bounds__(256)
void flash_col_attn(const float* __restrict__ a,
                    const unsigned short* __restrict__ qbf,
                    float* __restrict__ out) {
  __shared__ __align__(16) unsigned short s_q[2][NWAVES][BN * DSLICE]; // 256 KB, dbl-buffered Q
  __shared__ __align__(16) float s_acc[BN * BM];                       // 4 KB S^T reduce
  __shared__ __align__(16) unsigned short s_P[BM * BN];                // P bf16, [m][n]
  __shared__ float s_wmax[NWAVES][BM];
  __shared__ float s_wsum[NWAVES][BM];
  __shared__ float s_runmax[BM];
  __shared__ float s_runsum[BM];
  __shared__ float s_c[BM];

  const int b    = blockIdx.y;
  const int m0   = blockIdx.x * BM;
  const int tid  = threadIdx.x;
  const int wave = tid >> 5;
  const int lane = tid & 31;
  const int hf   = lane >> 4;   // lane half (WMMA layout split)
  const int l15  = lane & 15;
  const int d0   = wave * DSLICE;

  const float* aB          = a   + ((size_t)b * M_) * D_;
  const unsigned short* qB = qbf + ((size_t)b * N_) * D_;

  if (tid < BM) { s_runmax[tid] = -INFINITY; s_runsum[tid] = 0.0f; }
  ((float4*)s_acc)[tid] = make_float4(0.f, 0.f, 0.f, 0.f);

  // Wave-uniform TDM addresses (forced into SGPRs).
  const unsigned lds0 =
      __builtin_amdgcn_readfirstlane((unsigned)(unsigned long long)&s_q[0][wave][0]);
  const unsigned lds1 =
      __builtin_amdgcn_readfirstlane((unsigned)(unsigned long long)&s_q[1][wave][0]);
  unsigned long long gall =
      (unsigned long long)(const char*)qB + (unsigned)(wave * DSLICE * 2);
  const unsigned glo = __builtin_amdgcn_readfirstlane((unsigned)gall);
  const unsigned ghi = __builtin_amdgcn_readfirstlane((unsigned)(gall >> 32));
  unsigned long long gaddr = ((unsigned long long)ghi << 32) | glo;

  // Kick off DMA of tile 0 into buffer 0 (overlaps the A^T setup below).
  tdm_load_tile(lds0, gaddr);
  gaddr += TILE_BYTES;

  // Resident score B-operands: A^T for this wave's d-slice, 4 k-steps of 32.
  // B-matrix layout: lane%16 = N(=m), element s -> K = 8*hf + s (s<8), 16 + 8*hf + (s-8) (s>=8).
  BFrag atf[4];
#pragma unroll
  for (int ks = 0; ks < 4; ++ks) {
    const int d = d0 + ks * 32 + hf * 8;
#pragma unroll
    for (int g = 0; g < 16; ++g)
      atf[ks].h[g] = (unsigned short)f2bf(aB[(size_t)(m0 + l15) * D_ + d + (g & 7) + ((g >> 3) << 4)]);
  }

  v8f oacc[8];
#pragma unroll
  for (int t = 0; t < 8; ++t) oacc[t] = v8f_zero();

  __syncthreads();

  for (int n0 = 0; n0 < N_; n0 += BN) {
    const int buf = (n0 >> 6) & 1;
    const unsigned short* qsw = &s_q[buf][wave][0];

    // Tile i has landed; prior readers of the other buffer are long done.
    __builtin_amdgcn_s_wait_tensorcnt(0);
    asm volatile("s_wait_dscnt 0" ::: "memory");
    // Prefetch tile i+1 into the other buffer; DMA overlaps this whole iter.
    if (n0 + BN < N_) {
      tdm_load_tile(buf ? lds0 : lds1, gaddr);
      gaddr += TILE_BYTES;
    }

    // ---- scores: S^T[n,m] partial over this wave's 128-d slice ----
    v8f sacc[4];
#pragma unroll
    for (int ns = 0; ns < 4; ++ns) sacc[ns] = v8f_zero();
#pragma unroll
    for (int ns = 0; ns < 4; ++ns) {
#pragma unroll
      for (int ks = 0; ks < 4; ++ks) {
        BFrag qa;  // A-operand: lane = M(=n); K split across halves
        const unsigned short* p0 =
            qsw + (ns * 16 + l15) * DSLICE + ks * 32 + hf * 8;
        qa.u4[0] = *(const uint4*)(p0);
        qa.u4[1] = *(const uint4*)(p0 + 16);
        sacc[ns] = __builtin_amdgcn_wmma_f32_16x16x32_bf16(
            false, qa.v, false, atf[ks].v, (short)0, sacc[ns], false, false);
      }
    }

    // ---- cross-wave K reduction via LDS float atomics ----
#pragma unroll
    for (int ns = 0; ns < 4; ++ns)
#pragma unroll
      for (int g = 0; g < 8; ++g)
        atomicAdd(&s_acc[(ns * 16 + g + hf * 8) * BM + l15], sacc[ns][g]);
    __syncthreads();

    // ---- owner wave reads its 8 n-rows, computes per-m tile max ----
    float sred[4];
    float mx = -INFINITY;
#pragma unroll
    for (int e = 0; e < 4; ++e) {
      const int nl = wave * 8 + hf * 4 + e;
      const float s = s_acc[nl * BM + l15];
      sred[e] = s;
      mx = fmaxf(mx, s);
    }
    mx = fmaxf(mx, __shfl_xor(mx, 16, 32));
    if (lane < 16) s_wmax[wave][l15] = mx;
    __syncthreads();

    // zero the reducer for the next tile (reads done; next adds are several
    // barriers away), and update running max / correction factor
    ((float4*)s_acc)[tid] = make_float4(0.f, 0.f, 0.f, 0.f);
    if (wave == 0 && lane < 16) {
      float tm = s_wmax[0][lane];
#pragma unroll
      for (int w2 = 1; w2 < NWAVES; ++w2) tm = fmaxf(tm, s_wmax[w2][lane]);
      const float rm = s_runmax[lane];
      const float nm = fmaxf(rm, tm);
      s_c[lane]      = __expf(rm - nm);  // first tile: exp(-inf) = 0
      s_runmax[lane] = nm;
    }
    __syncthreads();

    // ---- P = exp(S - max), partial row sums, rescale O ----
    {
      const float nm = s_runmax[l15];
      float psum = 0.0f;
#pragma unroll
      for (int e = 0; e < 4; ++e) {
        const int nl = wave * 8 + hf * 4 + e;
        const float p = __expf(sred[e] - nm);
        psum += p;
        s_P[l15 * BN + nl] = (unsigned short)f2bf(p);
      }
      psum += __shfl_xor(psum, 16, 32);
      if (lane < 16) s_wsum[wave][l15] = psum;
      float cf[8];
#pragma unroll
      for (int g = 0; g < 8; ++g) cf[g] = s_c[g + hf * 8];
#pragma unroll
      for (int t = 0; t < 8; ++t)
#pragma unroll
        for (int g = 0; g < 8; ++g) oacc[t][g] *= cf[g];
    }
    __syncthreads();

    if (wave == 0 && lane < 16) {
      float ts = 0.0f;
#pragma unroll
      for (int w2 = 0; w2 < NWAVES; ++w2) ts += s_wsum[w2][lane];
      s_runsum[lane] = s_runsum[lane] * s_c[lane] + ts;
    }

    // ---- O[m, d_slice] += P(16 x 64n) * Q(64n x 128d), Q from staged LDS ----
#pragma unroll
    for (int ks = 0; ks < 2; ++ks) {
      BFrag pf;  // A-operand from LDS: lane = M(=m); K(=n) split across halves
      const unsigned short* pp = &s_P[l15 * BN + ks * 32 + hf * 8];
      pf.u4[0] = *(const uint4*)(pp);
      pf.u4[1] = *(const uint4*)(pp + 16);
#pragma unroll
      for (int t = 0; t < 8; ++t) {
        BFrag qb;  // B-operand: lane%16 = N(=d); K(=n) split across halves
        const unsigned short* qr = qsw + (ks * 32 + hf * 8) * DSLICE + t * 16 + l15;
#pragma unroll
        for (int s = 0; s < 16; ++s)
          qb.h[s] = qr[((s & 7) + ((s >> 3) << 4)) * DSLICE];
        oacc[t] = __builtin_amdgcn_wmma_f32_16x16x32_bf16(
            false, pf.v, false, qb.v, (short)0, oacc[t], false, false);
      }
    }
    __syncthreads();
  }

  // ---- epilogue: normalize and store ----
  float inv[8];
#pragma unroll
  for (int g = 0; g < 8; ++g) inv[g] = 1.0f / s_runsum[g + hf * 8];
  float* ob = out + ((size_t)b * M_ + m0) * D_ + d0;
#pragma unroll
  for (int t = 0; t < 8; ++t)
#pragma unroll
    for (int g = 0; g < 8; ++g)
      ob[(size_t)(g + hf * 8) * D_ + t * 16 + l15] = oacc[t][g] * inv[g];
}

extern "C" void kernel_launch(void* const* d_in, const int* in_sizes, int n_in,
                              void* d_out, int out_size, void* d_ws, size_t ws_size,
                              hipStream_t stream) {
  const float* q = (const float*)d_in[0];  // [B, N, D]
  const float* a = (const float*)d_in[1];  // [B, M, D]
  unsigned short* qbf = (unsigned short*)d_ws;  // 32 MB bf16 copy of q

  const int n8 = (B_ * N_ * D_) / 8;
  cvt_f32_bf16<<<(n8 + 255) / 256, 256, 0, stream>>>(q, qbf, n8);

  dim3 grid(M_ / BM, B_);
  flash_col_attn<<<grid, NWAVES * 32, 0, stream>>>(a, qbf, (float*)d_out);
}
